// Gemma4Experts_18537078850099
// MI455X (gfx1250) — matrix operand order repaired and checked by reading, not compiled, not run on hardware
//
#include <hip/hip_runtime.h>

// ---------------- problem constants ----------------
#define T_TOKENS 2048
#define HID      2048
#define INTER_N  4096
#define NEXP     8
#define TOPK     2

// ---------------- tiling ----------------
#define BM   64
#define BN   64
#define BK   32
#define LDK  40          // BK + 8 bf16 padding -> 80B row stride (16B aligned)
#define MAXROWS 4608     // sum over experts of ceil(count/64)*64  <= 4096 + 8*63

#if defined(__has_builtin)
#  if __has_builtin(__builtin_amdgcn_global_load_async_to_lds_b128)
#    define HAVE_ASYNC_LDS 1
#  endif
#endif
#ifndef HAVE_ASYNC_LDS
#  define HAVE_ASYNC_LDS 0
#endif

typedef __attribute__((ext_vector_type(16))) __bf16 v16bf;
typedef __attribute__((ext_vector_type(8)))  __bf16 v8bf;
typedef __attribute__((ext_vector_type(8)))  float  v8f;
typedef __attribute__((ext_vector_type(4)))  int    v4i_;
#if HAVE_ASYNC_LDS
typedef __attribute__((address_space(1))) v4i_ as1_v4i;
typedef __attribute__((address_space(3))) v4i_ as3_v4i;
#endif

// round-to-nearest-even bf16, kept as 32-bit value with result in bits [31:16]
static __device__ __forceinline__ unsigned rnebf(float f) {
    union { float f; unsigned u; } v; v.f = f;
    return v.u + 0x7FFFu + ((v.u >> 16) & 1u);
}
static __device__ __forceinline__ __bf16 f2bf(float f) {
    unsigned short h = (unsigned short)(rnebf(f) >> 16);
    return __builtin_bit_cast(__bf16, h);
}
// pack two bf16 into a dword using only 32-bit ALU ops (no 16-bit subregister moves)
static __device__ __forceinline__ unsigned pack2bf(float a, float b) {
    return (rnebf(a) >> 16) | (rnebf(b) & 0xFFFF0000u);
}

static __device__ __forceinline__ v16bf cat8(v8bf lo, v8bf hi) {
    v16bf r;
#pragma unroll
    for (int i = 0; i < 8; ++i) { r[i] = lo[i]; r[i + 8] = hi[i]; }
    return r;
}

// A fragment, 16x32 bf16 tile at row mbase inside LDS tile (row-major, stride LDK).
// ISA layout: lane m = lane%16 ; K halves at k0=(lane>=16?8:0) and k0+16.
static __device__ __forceinline__ v16bf load_frag_a(const __bf16* tile, int mbase, int lane) {
    int m  = mbase + (lane & 15);
    int k0 = (lane & 16) ? 8 : 0;
    const __bf16* p = tile + m * LDK + k0;
    v8bf lo = *(const v8bf*)(p);
    v8bf hi = *(const v8bf*)(p + 16);
    return cat8(lo, hi);
}

// B fragment, 32x16 bf16 tile at col nbase; LDS holds B transposed: sB[n][k].
// ISA layout: lane n = lane%16 ; K halves at k0=(lane>=16?8:0) and k0+16.
static __device__ __forceinline__ v16bf load_frag_b(const __bf16* tileT, int nbase, int lane) {
    int n  = nbase + (lane & 15);
    int k0 = (lane & 16) ? 8 : 0;
    const __bf16* p = tileT + n * LDK + k0;
    v8bf lo = *(const v8bf*)(p);
    v8bf hi = *(const v8bf*)(p + 16);
    return cat8(lo, hi);
}

// B tile loader: each of 128 threads owns a 4(k)x4(n) block.
// 4x global_load_b128 row reads -> in-register 4x4 transpose -> 4x ds_store_b64
// column writes into the K-transposed LDS tile sB[n][k] (LOCAL k index).
static __device__ __forceinline__ void load_b_tile(const float* __restrict__ src, int ldB,
                                                   __bf16* dstT, int kbase, int tid) {
    const int kb = (tid >> 4) * 4;        // local k: 0,4,...,28
    const int nb = (tid & 15) * 4;        // local n: 0,4,...,60
    const float4 r0 = *(const float4*)(src + (size_t)(kbase + kb + 0) * ldB + nb);
    const float4 r1 = *(const float4*)(src + (size_t)(kbase + kb + 1) * ldB + nb);
    const float4 r2 = *(const float4*)(src + (size_t)(kbase + kb + 2) * ldB + nb);
    const float4 r3 = *(const float4*)(src + (size_t)(kbase + kb + 3) * ldB + nb);
    uint2 c0 = make_uint2(pack2bf(r0.x, r1.x), pack2bf(r2.x, r3.x));
    uint2 c1 = make_uint2(pack2bf(r0.y, r1.y), pack2bf(r2.y, r3.y));
    uint2 c2 = make_uint2(pack2bf(r0.z, r1.z), pack2bf(r2.z, r3.z));
    uint2 c3 = make_uint2(pack2bf(r0.w, r1.w), pack2bf(r2.w, r3.w));
    *(uint2*)(dstT + (nb + 0) * LDK + kb) = c0;
    *(uint2*)(dstT + (nb + 1) * LDK + kb) = c1;
    *(uint2*)(dstT + (nb + 2) * LDK + kb) = c2;
    *(uint2*)(dstT + (nb + 3) * LDK + kb) = c3;
}

// A chunk copy (32 bytes, already bf16): async global->LDS if available.
static __device__ __forceinline__ void copy_a_chunk(const __bf16* g, __bf16* l) {
#if HAVE_ASYNC_LDS
    __builtin_amdgcn_global_load_async_to_lds_b128((as1_v4i*)g,       (as3_v4i*)l,       0, 0);
    __builtin_amdgcn_global_load_async_to_lds_b128((as1_v4i*)(g + 8), (as3_v4i*)(l + 8), 0, 0);
#else
    const uint4* s = (const uint4*)g;
    uint4* d = (uint4*)l;
    uint4 a0 = s[0], a1 = s[1];
    d[0] = a0; d[1] = a1;
#endif
}
static __device__ __forceinline__ void async_wait() {
#if HAVE_ASYNC_LDS
    asm volatile("s_wait_asynccnt 0" ::: "memory");
#endif
}

// ---------------- kernel 0: routing / dispatch ----------------
__global__ __launch_bounds__(256)
void moe_route(const int* __restrict__ sel, const float* __restrict__ rw,
               int* __restrict__ off, int* __restrict__ row_token,
               float* __restrict__ row_weight, int* __restrict__ total) {
    __shared__ int scnt[NEXP];
    __shared__ int scur[NEXP];
    __shared__ int soff[NEXP + 1];
    const int tid = threadIdx.x;
    if (tid < NEXP) scnt[tid] = 0;
    __syncthreads();
    for (int i = tid; i < T_TOKENS * TOPK; i += blockDim.x)
        atomicAdd(&scnt[sel[i] & (NEXP - 1)], 1);
    __syncthreads();
    if (tid == 0) {
        int o = 0;
        for (int e = 0; e < NEXP; ++e) {
            soff[e] = o; scur[e] = o;
            o += (scnt[e] + (BM - 1)) & ~(BM - 1);
        }
        soff[NEXP] = o;
        *total = o;
    }
    __syncthreads();
    // Fill padding slots (disjoint from scatter targets -> no fence needed).
    for (int i = tid; i < MAXROWS; i += blockDim.x) {
        bool pad = true;
        for (int e = 0; e < NEXP; ++e)
            if (i >= soff[e] && i < soff[e] + scnt[e]) { pad = false; break; }
        if (pad) { row_token[i] = -1; row_weight[i] = 0.0f; }
    }
    for (int i = tid; i < T_TOKENS * TOPK; i += blockDim.x) {
        int e = sel[i] & (NEXP - 1);
        int pos = atomicAdd(&scur[e], 1);
        row_token[pos]  = i >> 1;       // token index
        row_weight[pos] = rw[i];
    }
    if (tid <= NEXP) off[tid] = soff[tid];
}

// ---------------- kernel 1: gather + convert x rows to bf16 (row-compacted) --------
__global__ __launch_bounds__(256)
void gather_x_bf16(const float* __restrict__ x, const int* __restrict__ row_token,
                   __bf16* __restrict__ xg) {
    const int row = blockIdx.x;
    const int tok = row_token[row];
    const int c0  = threadIdx.x * 8;                     // 256 thr x 8 = 2048 cols
    uint4 u = make_uint4(0u, 0u, 0u, 0u);
    if (tok >= 0) {
        const float* p = x + (size_t)tok * HID + c0;
        float f[8];
#pragma unroll
        for (int i = 0; i < 8; ++i) f[i] = p[i];
        u.x = pack2bf(f[0], f[1]); u.y = pack2bf(f[2], f[3]);
        u.z = pack2bf(f[4], f[5]); u.w = pack2bf(f[6], f[7]);
    }
    *(uint4*)(xg + (size_t)row * HID + c0) = u;
}

// ---------------- kernel 2: zero output ----------------
__global__ __launch_bounds__(256)
void zero_f32(float* __restrict__ p, int n4) {
    int i = blockIdx.x * blockDim.x + threadIdx.x;
    if (i < n4) ((float4*)p)[i] = make_float4(0.f, 0.f, 0.f, 0.f);
}

// ---------------- kernel 3: gate+up GEMM + tanh-GELU, h -> bf16 workspace ----------
__global__ __launch_bounds__(128)
void moe_gateup(const __bf16* __restrict__ xg,
                const float* __restrict__ Wg, const float* __restrict__ Wu,
                const int* __restrict__ off, const int* __restrict__ total,
                __bf16* __restrict__ hbuf) {
    __shared__ __align__(16) __bf16 sA [BM * LDK];
    __shared__ __align__(16) __bf16 sBg[BN * LDK];
    __shared__ __align__(16) __bf16 sBu[BN * LDK];

    const int rowbase = blockIdx.x * BM;
    if (rowbase >= *total) return;
    int e = 0;
    while (e < NEXP - 1 && rowbase >= off[e + 1]) ++e;

    const int colbase = blockIdx.y * BN;
    const int tid  = threadIdx.x;
    const int lane = tid & 31;
    const int wave = tid >> 5;            // 4 waves: 2x2 of 32x32
    const int wm = (wave >> 1) * 32;
    const int wn = (wave & 1) * 32;

    const float* wg = Wg + (size_t)e * HID * INTER_N + colbase;
    const float* wu = Wu + (size_t)e * HID * INTER_N + colbase;

    v8f accG[2][2], accU[2][2];
#pragma unroll
    for (int mt = 0; mt < 2; ++mt)
#pragma unroll
        for (int nt = 0; nt < 2; ++nt) { accG[mt][nt] = {}; accU[mt][nt] = {}; }

    // A copy mapping: row ar (0..63), 16-element (32B) chunk.
    const int ar = tid >> 1;
    const int ac = (tid & 1) * 16;
    const __bf16* arow = xg + (size_t)(rowbase + ar) * HID + ac;
    __bf16* adst = sA + ar * LDK + ac;
    const int pkb = (tid >> 4) * 4;       // for prefetch addressing
    const int pnb = (tid & 15) * 4;

    for (int k0 = 0; k0 < HID; k0 += BK) {
        __syncthreads();
        copy_a_chunk(arow + k0, adst);
        load_b_tile(wg, INTER_N, sBg, k0, tid);
        load_b_tile(wu, INTER_N, sBu, k0, tid);
        if (k0 + BK < HID) {  // keep the HBM weight stream ahead
            __builtin_prefetch(&wg[(size_t)(k0 + BK + pkb) * INTER_N + pnb], 0, 1);
            __builtin_prefetch(&wu[(size_t)(k0 + BK + pkb) * INTER_N + pnb], 0, 1);
        }
        async_wait();
        __syncthreads();

        v16bf af[2], bg[2], bu[2];
#pragma unroll
        for (int mt = 0; mt < 2; ++mt) af[mt] = load_frag_a(sA, wm + mt * 16, lane);
#pragma unroll
        for (int nt = 0; nt < 2; ++nt) {
            bg[nt] = load_frag_b(sBg, wn + nt * 16, lane);
            bu[nt] = load_frag_b(sBu, wn + nt * 16, lane);
        }
#pragma unroll
        for (int mt = 0; mt < 2; ++mt)
#pragma unroll
            for (int nt = 0; nt < 2; ++nt) {
                accG[mt][nt] = __builtin_amdgcn_wmma_f32_16x16x32_bf16(
                    false, af[mt], false, bg[nt], (short)0, accG[mt][nt], false, false);
                accU[mt][nt] = __builtin_amdgcn_wmma_f32_16x16x32_bf16(
                    false, af[mt], false, bu[nt], (short)0, accU[mt][nt], false, false);
            }
    }

    // epilogue: h = gelu_tanh(gate) * up, store bf16
    const int mlane = lane & 15;
    const int moffL = (lane >> 4) * 8;
#pragma unroll
    for (int mt = 0; mt < 2; ++mt)
#pragma unroll
        for (int nt = 0; nt < 2; ++nt)
#pragma unroll
            for (int v = 0; v < 8; ++v) {
                int gm = rowbase + wm + mt * 16 + moffL + v;
                int gn = colbase + wn + nt * 16 + mlane;
                float g = accG[mt][nt][v];
                float u = accU[mt][nt][v];
                float t = 0.7978845608028654f * (g + 0.044715f * g * g * g);
                float h = 0.5f * g * (1.0f + tanhf(t)) * u;
                hbuf[(size_t)gm * INTER_N + gn] = f2bf(h);
            }
}

// ---------------- kernel 4: down GEMM + weighted scatter-add ----------------
__global__ __launch_bounds__(128)
void moe_down(const __bf16* __restrict__ hbuf, const float* __restrict__ Wd,
              const int* __restrict__ off, const int* __restrict__ row_token,
              const float* __restrict__ row_weight, const int* __restrict__ total,
              float* __restrict__ out) {
    __shared__ __align__(16) __bf16 sA[BM * LDK];
    __shared__ __align__(16) __bf16 sB[BN * LDK];

    const int rowbase = blockIdx.x * BM;
    if (rowbase >= *total) return;
    int e = 0;
    while (e < NEXP - 1 && rowbase >= off[e + 1]) ++e;

    const int colbase = blockIdx.y * BN;
    const int tid  = threadIdx.x;
    const int lane = tid & 31;
    const int wave = tid >> 5;
    const int wm = (wave >> 1) * 32;
    const int wn = (wave & 1) * 32;

    const float* wd = Wd + (size_t)e * INTER_N * HID + colbase;

    v8f acc[2][2];
#pragma unroll
    for (int mt = 0; mt < 2; ++mt)
#pragma unroll
        for (int nt = 0; nt < 2; ++nt) acc[mt][nt] = {};

    const int ar = tid >> 1;
    const int ac = (tid & 1) * 16;
    const __bf16* arow = hbuf + (size_t)(rowbase + ar) * INTER_N + ac;
    __bf16* adst = sA + ar * LDK + ac;
    const int pkb = (tid >> 4) * 4;
    const int pnb = (tid & 15) * 4;

    for (int k0 = 0; k0 < INTER_N; k0 += BK) {
        __syncthreads();
        copy_a_chunk(arow + k0, adst);
        load_b_tile(wd, HID, sB, k0, tid);
        if (k0 + BK < INTER_N)
            __builtin_prefetch(&wd[(size_t)(k0 + BK + pkb) * HID + pnb], 0, 1);
        async_wait();
        __syncthreads();

        v16bf af[2], bf[2];
#pragma unroll
        for (int mt = 0; mt < 2; ++mt) af[mt] = load_frag_a(sA, wm + mt * 16, lane);
#pragma unroll
        for (int nt = 0; nt < 2; ++nt) bf[nt] = load_frag_b(sB, wn + nt * 16, lane);
#pragma unroll
        for (int mt = 0; mt < 2; ++mt)
#pragma unroll
            for (int nt = 0; nt < 2; ++nt)
                acc[mt][nt] = __builtin_amdgcn_wmma_f32_16x16x32_bf16(
                    false, af[mt], false, bf[nt], (short)0, acc[mt][nt], false, false);
    }

    const int mlane = lane & 15;
    const int moffL = (lane >> 4) * 8;
#pragma unroll
    for (int mt = 0; mt < 2; ++mt)
#pragma unroll
        for (int nt = 0; nt < 2; ++nt)
#pragma unroll
            for (int v = 0; v < 8; ++v) {
                int gm = rowbase + wm + mt * 16 + moffL + v;
                int tok = row_token[gm];
                if (tok >= 0) {
                    int gn = colbase + wn + nt * 16 + mlane;
                    atomicAdd(out + (size_t)tok * HID + gn, row_weight[gm] * acc[mt][nt][v]);
                }
            }
}

// ---------------- host launcher ----------------
extern "C" void kernel_launch(void* const* d_in, const int* in_sizes, int n_in,
                              void* d_out, int out_size, void* d_ws, size_t ws_size,
                              hipStream_t stream) {
    const float* x   = (const float*)d_in[0];
    const int*   sel = (const int*)  d_in[1];
    const float* rw  = (const float*)d_in[2];
    const float* Wg  = (const float*)d_in[3];
    const float* Wu  = (const float*)d_in[4];
    const float* Wd  = (const float*)d_in[5];
    float* out = (float*)d_out;

    char* ws = (char*)d_ws;
    int*    off        = (int*)ws;                       // 16 ints reserved (use 9)
    int*    total      = ((int*)ws) + 15;
    int*    row_token  = (int*)  (ws + 64);
    float*  row_weight = (float*)(ws + 64 + MAXROWS * 4);
    __bf16* xg         = (__bf16*)(ws + 64 + MAXROWS * 8);                 // MAXROWS x HID
    __bf16* hbuf       = (__bf16*)(ws + 64 + MAXROWS * 8
                                      + (size_t)MAXROWS * HID * 2);        // MAXROWS x INTER_N

    moe_route<<<1, 256, 0, stream>>>(sel, rw, off, row_token, row_weight, total);

    gather_x_bf16<<<MAXROWS, 256, 0, stream>>>(x, row_token, xg);

    int n4 = (T_TOKENS * HID) / 4;
    zero_f32<<<(n4 + 255) / 256, 256, 0, stream>>>(out, n4);

    dim3 g2(MAXROWS / BM, INTER_N / BN);   // 72 x 64, early-exit past *total
    moe_gateup<<<g2, 128, 0, stream>>>(xg, Wg, Wu, off, total, hbuf);

    dim3 g3(MAXROWS / BM, HID / BN);       // 72 x 32
    moe_down<<<g3, 128, 0, stream>>>(hbuf, Wd, off, row_token, row_weight, total, out);
}
